// NonIdenticalQuadLayer_48756468744441
// MI455X (gfx1250) — matrix operand order repaired and checked by reading, not compiled, not run on hardware
//
#include <hip/hip_runtime.h>

typedef __attribute__((ext_vector_type(16))) __bf16 v16bf;
typedef __attribute__((ext_vector_type(8)))  float  v8f;

namespace {
constexpr int kBS   = 1024;
constexpr int kN1   = 512;
constexpr int kN2   = 512;
constexpr int kOUT  = 1024;
constexpr int kKlin = kN1 + kN2;          // 1024 linear features [x|y]
constexpr int kKq   = kN1 * kN2;          // 262144 outer-product features
constexpr int kWst  = kKq + kKlin;        // 263168 = W row stride
constexpr int TM    = 128;                // WG tile M (batch rows)
constexpr int TN    = 64;                 // WG tile N (output features)
constexpr int LDSK  = kKlin;              // 1024 bf16 per LDS row ([x|y])
}

union Frag { unsigned u[8]; v16bf v; };

// single-instruction RNE pack of two f32 -> packed bf16 (exists on CDNA3/4;
// probing whether gfx1250 kept it — fallback is add+perm if this fails)
__device__ __forceinline__ unsigned cvt_pk_bf16(float lo, float hi) {
  unsigned d;
  asm("v_cvt_pk_bf16_f32 %0, %1, %2" : "=v"(d) : "v"(lo), "v"(hi));
  return d;
}

__device__ __forceinline__ unsigned pk_mul_bf16(unsigned a, unsigned b) {
  unsigned d;
  asm("v_pk_mul_bf16 %0, %1, %2" : "=v"(d) : "v"(a), "v"(b));
  return d;
}

// B fragment: lane = column (l16), half-wave selects K base (hi*8); each lane
// reads f32 K = {0..7} and {16..23} of one W row and packs pairs -> 8 VGPRs.
__device__ __forceinline__ void load_b_frag(const float* __restrict__ p, Frag& f) {
  float4 q0 = *(const float4*)(p + 0);
  float4 q1 = *(const float4*)(p + 4);
  float4 q2 = *(const float4*)(p + 16);
  float4 q3 = *(const float4*)(p + 20);
  f.u[0] = cvt_pk_bf16(q0.x, q0.y); f.u[1] = cvt_pk_bf16(q0.z, q0.w);
  f.u[2] = cvt_pk_bf16(q1.x, q1.y); f.u[3] = cvt_pk_bf16(q1.z, q1.w);
  f.u[4] = cvt_pk_bf16(q2.x, q2.y); f.u[5] = cvt_pk_bf16(q2.z, q2.w);
  f.u[6] = cvt_pk_bf16(q3.x, q3.y); f.u[7] = cvt_pk_bf16(q3.z, q3.w);
}

// A-layout slots straight from LDS. CDNA5 16-bit A layout per lane:
// VGPR pairs hold K = hi*8 + {0..3}, {4..7}, {16..19}, {20..23}.
__device__ __forceinline__ void load_a_slots(const unsigned short* cat, int base, Frag& f) {
  uint2 g0 = *(const uint2*)&cat[base +  0];
  uint2 g1 = *(const uint2*)&cat[base +  4];
  uint2 g2 = *(const uint2*)&cat[base + 16];
  uint2 g3 = *(const uint2*)&cat[base + 20];
  f.u[0] = g0.x; f.u[1] = g0.y; f.u[2] = g1.x; f.u[3] = g1.y;
  f.u[4] = g2.x; f.u[5] = g2.y; f.u[6] = g3.x; f.u[7] = g3.y;
}

#define WMMA_BF16(A, B, C) \
  __builtin_amdgcn_wmma_f32_16x16x32_bf16(false, (A), false, (B), (short)0, (C), false, false)

__global__ __launch_bounds__(256, 1) void quad_gemm_kernel(
    const float* __restrict__ x, const float* __restrict__ y,
    const float* __restrict__ W, float* __restrict__ out)
{
  extern __shared__ unsigned short cat[];   // [TM][LDSK] bf16 = 256 KB
  const int tid = threadIdx.x;
  const int m0  = blockIdx.y * TM;
  const int n0  = blockIdx.x * TN;

  // ---- stage [x | y] tile to LDS as bf16 (pairs never cross the 512 boundary) ----
  for (int f = tid * 2; f < TM * LDSK; f += 256 * 2) {
    const int row = f >> 10;
    const int c   = f & 1023;
    const float* src = (c < kN1) ? (x + (size_t)(m0 + row) * kN1 + c)
                                 : (y + (size_t)(m0 + row) * kN2 + (c - kN1));
    float2 v = *(const float2*)src;
    *(unsigned*)&cat[f] = cvt_pk_bf16(v.x, v.y);
  }
  __syncthreads();

  const int lane = tid & 31;
  const int w    = tid >> 5;
  const int wm   = (w & 3) * 32;       // 4 waves along M
  const int wn   = (w >> 2) * 32;      // 2 waves along N
  const int l16  = lane & 15;
  const int hi   = lane >> 4;
  const int khi8 = hi * 8;

  v8f acc[2][2] = {};

  const float* wrow0 = W + (size_t)(n0 + wn +  0 + l16) * kWst + hi * 8;
  const float* wrow1 = W + (size_t)(n0 + wn + 16 + l16) * kWst + hi * 8;

  const int rbase0 = (wm +  0 + l16) * LDSK;
  const int rbase1 = (wm + 16 + l16) * LDSK;

  // ---- linear term: K = 0..1023, A read directly from [x|y] tile ----
  for (int kq = 0; kq < kKlin; kq += 32) {
    Frag a0, a1, b0, b1;
    load_a_slots(cat, rbase0 + kq + khi8, a0);
    load_a_slots(cat, rbase1 + kq + khi8, a1);
    load_b_frag(wrow0 + kq, b0);
    load_b_frag(wrow1 + kq, b1);
    acc[0][0] = WMMA_BF16(a0.v, b0.v, acc[0][0]);
    acc[0][1] = WMMA_BF16(a0.v, b1.v, acc[0][1]);
    acc[1][0] = WMMA_BF16(a1.v, b0.v, acc[1][0]);
    acc[1][1] = WMMA_BF16(a1.v, b1.v, acc[1][1]);
  }

  // ---- quadratic term: A[b, i*N2+j] = x[b,i]*y[b,j], built from registers ----
  // Outer loop over 16 j-chunks: the lane's 16 y values (A-layout slots) are
  // hoisted into registers; inner loop over i needs only 2 ds_load_u16 (x) and
  // 16 v_pk_mul_bf16 per 4 WMMAs. W is swept at stride N2*4B = 2KB, each visit
  // consuming one full 128B line -> exactly 1x HBM traffic for W.
  const float* wq0 = wrow0 + kKlin;
  const float* wq1 = wrow1 + kKlin;
  for (int j0 = 0; j0 < kN2; j0 += 32) {
    Frag y0, y1;
    load_a_slots(cat, rbase0 + kN1 + j0 + khi8, y0);
    load_a_slots(cat, rbase1 + kN1 + j0 + khi8, y1);
    const float* wb0 = wq0 + j0;
    const float* wb1 = wq1 + j0;
    #pragma unroll 2
    for (int i = 0; i < kN1; ++i) {
      unsigned xd0 = cat[rbase0 + i]; xd0 |= xd0 << 16;   // dup bf16 scalar
      unsigned xd1 = cat[rbase1 + i]; xd1 |= xd1 << 16;
      Frag a0, a1, b0, b1;
      #pragma unroll
      for (int p = 0; p < 8; ++p) {
        a0.u[p] = pk_mul_bf16(xd0, y0.u[p]);
        a1.u[p] = pk_mul_bf16(xd1, y1.u[p]);
      }
      load_b_frag(wb0 + (size_t)i * kN2, b0);
      load_b_frag(wb1 + (size_t)i * kN2, b1);
      __builtin_prefetch(wb0 + (size_t)(i + 8) * kN2, 0, 3);  // 16KB ahead, near-cache
      __builtin_prefetch(wb1 + (size_t)(i + 8) * kN2, 0, 3);
      acc[0][0] = WMMA_BF16(a0.v, b0.v, acc[0][0]);
      acc[0][1] = WMMA_BF16(a0.v, b1.v, acc[0][1]);
      acc[1][0] = WMMA_BF16(a1.v, b0.v, acc[1][0]);
      acc[1][1] = WMMA_BF16(a1.v, b1.v, acc[1][1]);
    }
  }

  // ---- store: f32 C/D layout (VGPR r -> row hi*8+r, col l16), no atomics ----
  #pragma unroll
  for (int s = 0; s < 2; ++s)
    #pragma unroll
    for (int t = 0; t < 2; ++t) {
      const int col = n0 + wn + t * 16 + l16;
      const int row = m0 + wm + s * 16 + hi * 8;
      float* o = out + (size_t)row * kOUT + col;
      #pragma unroll
      for (int r = 0; r < 8; ++r)
        o[(size_t)r * kOUT] = acc[s][t][r];
    }
}

extern "C" void kernel_launch(void* const* d_in, const int* in_sizes, int n_in,
                              void* d_out, int out_size, void* d_ws, size_t ws_size,
                              hipStream_t stream) {
  const float* x = (const float*)d_in[0];   // [1024, 512]
  const float* y = (const float*)d_in[1];   // [1024, 512]
  const float* W = (const float*)d_in[2];   // [1024, 263168]
  float* out = (float*)d_out;               // [1024, 1024]
  dim3 grid(kOUT / TN, kBS / TM);           // (16, 8) = 128 WGs
  size_t lds_bytes = (size_t)TM * LDSK * sizeof(unsigned short);  // 256 KB
  quad_gemm_kernel<<<grid, dim3(256), lds_bytes, stream>>>(x, y, W, out);
}
